// GraphNet_6734508720674
// MI455X (gfx1250) — matrix operand order repaired and checked by reading, not compiled, not run on hardware
//
#include <hip/hip_runtime.h>
#include <hip/hip_bf16.h>
#include <stdint.h>

typedef __bf16 bf16;
typedef __attribute__((ext_vector_type(16))) __bf16 v16bf;
typedef __attribute__((ext_vector_type(8)))  __bf16 v8bf;
typedef __attribute__((ext_vector_type(8)))  float  v8f;
typedef __attribute__((ext_vector_type(4)))  unsigned int v4u;
typedef __attribute__((ext_vector_type(8)))  unsigned int v8u;

#define N_NODES 50000
#define F_DIM   128     // F_IN == H == 128
#define MH_DIM  256
#define N_EDGES 800000
#define WROW    264     // padded LDS row stride in bf16 (512B row + 16B TDM pad)

// ---------------------------------------------------------------------------
// elementwise helpers
// ---------------------------------------------------------------------------
__global__ void k_zero_f32(float* __restrict__ p, int n) {
    int i = blockIdx.x * blockDim.x + threadIdx.x;
    if (i < n) p[i] = 0.0f;
}

__global__ void k_cvt_bf16(const float* __restrict__ s, bf16* __restrict__ d, int n) {
    int i = blockIdx.x * blockDim.x + threadIdx.x;
    if (i < n) d[i] = (bf16)s[i];
}

// ---------------------------------------------------------------------------
// scatter-sum of f32 features + degree count (layer 1)
// ---------------------------------------------------------------------------
__global__ void k_scatter_f32(const float* __restrict__ x,
                              const long long* __restrict__ srcI,
                              const long long* __restrict__ dstI,
                              float* __restrict__ agg, float* __restrict__ cnt) {
    int t = blockIdx.x * blockDim.x + threadIdx.x;   // E * 32 threads
    int e = t >> 5, c = t & 31;
    if (e >= N_EDGES) return;
    int s = (int)srcI[e], d = (int)dstI[e];
    float4 v = *(const float4*)(x + s * F_DIM + c * 4);
    float* a = agg + d * F_DIM + c * 4;
    atomicAdd(a + 0, v.x); atomicAdd(a + 1, v.y);
    atomicAdd(a + 2, v.z); atomicAdd(a + 3, v.w);
    if (c == 0) atomicAdd(cnt + d, 1.0f);
}

// scatter-sum of bf16 features (layer 2)
__global__ void k_scatter_bf16(const bf16* __restrict__ h,
                               const long long* __restrict__ srcI,
                               const long long* __restrict__ dstI,
                               float* __restrict__ agg) {
    int t = blockIdx.x * blockDim.x + threadIdx.x;
    int e = t >> 5, c = t & 31;
    if (e >= N_EDGES) return;
    int s = (int)srcI[e], d = (int)dstI[e];
    const bf16* p = h + s * F_DIM + c * 4;
    float* a = agg + d * F_DIM + c * 4;
    atomicAdd(a + 0, (float)p[0]); atomicAdd(a + 1, (float)p[1]);
    atomicAdd(a + 2, (float)p[2]); atomicAdd(a + 3, (float)p[3]);
}

// agg / max(cnt,1) -> bf16
__global__ void k_mean_bf16(const float* __restrict__ agg, const float* __restrict__ cnt,
                            bf16* __restrict__ aggb) {
    int i = blockIdx.x * blockDim.x + threadIdx.x;
    if (i >= N_NODES * F_DIM) return;
    float c = fmaxf(cnt[i >> 7], 1.0f);
    aggb[i] = (bf16)(agg[i] / c);
}

// ---------------------------------------------------------------------------
// SAGE layer: out = relu(aggb @ Wl^T + bias + xb @ Wr^T), bf16 WMMA, f32 acc
// one wave per 16x16 output tile; 25000 tiles
// ---------------------------------------------------------------------------
__global__ __launch_bounds__(256)
void k_sage_wmma(const bf16* __restrict__ aggb, const bf16* __restrict__ xb,
                 const bf16* __restrict__ Wl, const bf16* __restrict__ Wr,
                 const float* __restrict__ bias, bf16* __restrict__ out) {
    const int lane = threadIdx.x & 31;
    const int wid  = threadIdx.x >> 5;
    const int tile = blockIdx.x * 8 + wid;
    const int tm = tile >> 3;
    const int tn = tile & 7;
    const int r  = lane & 15;
    const int hf = lane >> 4;

    const bf16* aRow0 = aggb + (tm * 16 + r) * F_DIM;
    const bf16* aRow1 = xb   + (tm * 16 + r) * F_DIM;
    const bf16* bRow0 = Wl   + (tn * 16 + r) * F_DIM;
    const bf16* bRow1 = Wr   + (tn * 16 + r) * F_DIM;

    v8f c = {};
#pragma unroll
    for (int k0 = 0; k0 < F_DIM; k0 += 32) {
        v16bf a, b;
        *((v8bf*)&a)       = *(const v8bf*)(aRow0 + k0 + hf * 8);
        *(((v8bf*)&a) + 1) = *(const v8bf*)(aRow0 + k0 + 16 + hf * 8);
        *((v8bf*)&b)       = *(const v8bf*)(bRow0 + k0 + hf * 8);
        *(((v8bf*)&b) + 1) = *(const v8bf*)(bRow0 + k0 + 16 + hf * 8);
        c = __builtin_amdgcn_wmma_f32_16x16x32_bf16(false, a, false, b, (short)0, c, false, false);

        *((v8bf*)&a)       = *(const v8bf*)(aRow1 + k0 + hf * 8);
        *(((v8bf*)&a) + 1) = *(const v8bf*)(aRow1 + k0 + 16 + hf * 8);
        *((v8bf*)&b)       = *(const v8bf*)(bRow1 + k0 + hf * 8);
        *(((v8bf*)&b) + 1) = *(const v8bf*)(bRow1 + k0 + 16 + hf * 8);
        c = __builtin_amdgcn_wmma_f32_16x16x32_bf16(false, a, false, b, (short)0, c, false, false);
    }
    const int col = tn * 16 + r;
    const float bc = bias[col];
#pragma unroll
    for (int j = 0; j < 8; ++j) {
        int row = tm * 16 + j + 8 * hf;
        out[row * F_DIM + col] = (bf16)fmaxf(c[j] + bc, 0.0f);
    }
}

// ---------------------------------------------------------------------------
// Edge MLP: out[e] = relu([h_src|h_dst] @ Wm1^T + bm1) . wm2 + bm2
//  - Wm1 (bf16, 128KB) staged in LDS ONCE per workgroup via the Tensor Data
//    Mover (tensor_load_to_lds). The TDM pad feature inserts 16B per 512B row
//    -> 528B LDS row stride, bank-conflict-friendly for per-lane b128 reads.
//  - each wave owns 32 edges (two M-tiles); every B fragment read from LDS
//    feeds two WMMAs (halves LDS bandwidth demand).
//  supertiles = E/32 = 25000; 8 waves/block -> 3125 blocks
// ---------------------------------------------------------------------------
__global__ __launch_bounds__(256)
void k_edge_mlp(const bf16* __restrict__ hb,
                const long long* __restrict__ srcI, const long long* __restrict__ dstI,
                const bf16* __restrict__ Wm1b, const float* __restrict__ bm1,
                const float* __restrict__ wm2, const float* __restrict__ bm2,
                float* __restrict__ out) {
    extern __shared__ bf16 smem[];           // 256 x WROW bf16 = 132 KB

    // ---- TDM stage: one descriptor copies all of Wm1b with LDS padding ----
    if (threadIdx.x < 32) {
        unsigned lds_off = (unsigned)(uintptr_t)(&smem[0]);       // LDS byte addr
        unsigned long long ga = (unsigned long long)(uintptr_t)Wm1b;
        v4u g0;
        g0[0] = 1u;                                               // count=1
        g0[1] = lds_off;                                          // lds_addr
        g0[2] = (unsigned)(ga & 0xFFFFFFFFull);                   // global_addr lo
        g0[3] = (unsigned)((ga >> 32) & 0x1FFFFFFull) | (2u << 30); // addr hi | type=2
        v8u g1;
        // data_size=3 (8B units), pad_enable, pad_interval=6 (per 128 DWORDs=512B),
        // pad_amount=3 (4 DWORDs = 16B)
        g1[0] = (3u << 16) | (1u << 20) | (6u << 22) | (3u << 25);
        g1[1] = (16384u & 0xFFFFu) << 16;    // tensor_dim0[15:0] (131072B / 8)
        g1[2] = (16384u >> 16) | (1u << 16); // tensor_dim0[31:16] | tensor_dim1=1
        g1[3] = (16384u & 0xFFFFu) << 16;    // tile_dim0 = 16384 units
        g1[4] = 1u;                          // tile_dim1 = 1
        g1[5] = 16384u;                      // tensor_dim0_stride lo
        g1[6] = 0u;
        g1[7] = 0u;
        asm volatile("tensor_load_to_lds %0, %1" :: "s"(g0), "s"(g1) : "memory");
        __builtin_amdgcn_s_wait_tensorcnt(0);
    }
    __syncthreads();

    const int lane = threadIdx.x & 31;
    const int wid  = threadIdx.x >> 5;
    const int st   = blockIdx.x * 8 + wid;   // supertile of 32 edges
    const int r    = lane & 15;
    const int hf   = lane >> 4;

    const int e0 = st * 32 + r;
    const int e1 = e0 + 16;
    const int s0 = (int)srcI[e0], d0 = (int)dstI[e0];
    const int s1 = (int)srcI[e1], d1 = (int)dstI[e1];
    const bf16* s0Row = hb + s0 * F_DIM;
    const bf16* d0Row = hb + d0 * F_DIM;
    const bf16* s1Row = hb + s1 * F_DIM;
    const bf16* d1Row = hb + d1 * F_DIM;

    // A fragments for both M-tiles (K = 256): kk 0..3 from h_src, 4..7 from h_dst
    v16bf A0[8], A1[8];
#pragma unroll
    for (int kk = 0; kk < 8; ++kk) {
        const bf16* b0 = (kk < 4) ? s0Row : d0Row;
        const bf16* b1 = (kk < 4) ? s1Row : d1Row;
        int off = (kk & 3) * 32;
        *((v8bf*)&A0[kk])       = *(const v8bf*)(b0 + off + hf * 8);
        *(((v8bf*)&A0[kk]) + 1) = *(const v8bf*)(b0 + off + 16 + hf * 8);
        *((v8bf*)&A1[kk])       = *(const v8bf*)(b1 + off + hf * 8);
        *(((v8bf*)&A1[kk]) + 1) = *(const v8bf*)(b1 + off + 16 + hf * 8);
    }

    float acc0[8], acc1[8];
#pragma unroll
    for (int j = 0; j < 8; ++j) { acc0[j] = 0.0f; acc1[j] = 0.0f; }

    for (int tn = 0; tn < 16; ++tn) {
        const bf16* bRow = smem + (tn * 16 + r) * WROW;   // B: lane = col = row of Wm1
        v8f c0 = {}, c1 = {};
#pragma unroll
        for (int kk = 0; kk < 8; ++kk) {
            v16bf b;
            *((v8bf*)&b)       = *(const v8bf*)(bRow + kk * 32 + hf * 8);
            *(((v8bf*)&b) + 1) = *(const v8bf*)(bRow + kk * 32 + 16 + hf * 8);
            c0 = __builtin_amdgcn_wmma_f32_16x16x32_bf16(false, A0[kk], false, b, (short)0, c0, false, false);
            c1 = __builtin_amdgcn_wmma_f32_16x16x32_bf16(false, A1[kk], false, b, (short)0, c1, false, false);
        }
        const int col = tn * 16 + r;
        const float bb = bm1[col];
        const float w2 = wm2[col];
#pragma unroll
        for (int j = 0; j < 8; ++j) {
            acc0[j] += fmaxf(c0[j] + bb, 0.0f) * w2;
            acc1[j] += fmaxf(c1[j] + bb, 0.0f) * w2;
        }
    }

    // reduce over the 16 columns held by lanes sharing the same half
#pragma unroll
    for (int j = 0; j < 8; ++j) {
        float v0 = acc0[j], v1 = acc1[j];
        v0 += __shfl_xor(v0, 1, 32);  v1 += __shfl_xor(v1, 1, 32);
        v0 += __shfl_xor(v0, 2, 32);  v1 += __shfl_xor(v1, 2, 32);
        v0 += __shfl_xor(v0, 4, 32);  v1 += __shfl_xor(v1, 4, 32);
        v0 += __shfl_xor(v0, 8, 32);  v1 += __shfl_xor(v1, 8, 32);
        acc0[j] = v0; acc1[j] = v1;
    }
    if (r == 0) {
        const float b2 = bm2[0];
#pragma unroll
        for (int j = 0; j < 8; ++j) {
            out[st * 32 + j + 8 * hf]      = acc0[j] + b2;
            out[st * 32 + 16 + j + 8 * hf] = acc1[j] + b2;
        }
    }
}

// ---------------------------------------------------------------------------
// launch
// ---------------------------------------------------------------------------
static inline size_t alignup(size_t v) { return (v + 255) & ~(size_t)255; }

extern "C" void kernel_launch(void* const* d_in, const int* in_sizes, int n_in,
                              void* d_out, int out_size, void* d_ws, size_t ws_size,
                              hipStream_t stream) {
    (void)in_sizes; (void)n_in; (void)out_size; (void)ws_size;
    const float*     x    = (const float*)d_in[0];
    const long long* ei   = (const long long*)d_in[1];
    const float*     W1l  = (const float*)d_in[2];
    const float*     b1l  = (const float*)d_in[3];
    const float*     W1r  = (const float*)d_in[4];
    const float*     W2l  = (const float*)d_in[5];
    const float*     b2l  = (const float*)d_in[6];
    const float*     W2r  = (const float*)d_in[7];
    const float*     Wm1  = (const float*)d_in[8];
    const float*     bm1  = (const float*)d_in[9];
    const float*     Wm2  = (const float*)d_in[10];
    const float*     bm2  = (const float*)d_in[11];
    float* out = (float*)d_out;

    const long long* srcI = ei;
    const long long* dstI = ei + N_EDGES;

    char* p = (char*)d_ws;
    float* cnt  = (float*)p;  p += alignup((size_t)N_NODES * 4);
    float* agg  = (float*)p;  p += alignup((size_t)N_NODES * F_DIM * 4);
    bf16* xb    = (bf16*)p;   p += alignup((size_t)N_NODES * F_DIM * 2);
    bf16* aggb  = (bf16*)p;   p += alignup((size_t)N_NODES * F_DIM * 2);
    bf16* hb1   = (bf16*)p;   p += alignup((size_t)N_NODES * F_DIM * 2);
    bf16* hb2   = (bf16*)p;   p += alignup((size_t)N_NODES * F_DIM * 2);
    bf16* W1lb  = (bf16*)p;   p += alignup((size_t)F_DIM * F_DIM * 2);
    bf16* W1rb  = (bf16*)p;   p += alignup((size_t)F_DIM * F_DIM * 2);
    bf16* W2lb  = (bf16*)p;   p += alignup((size_t)F_DIM * F_DIM * 2);
    bf16* W2rb  = (bf16*)p;   p += alignup((size_t)F_DIM * F_DIM * 2);
    bf16* Wm1b  = (bf16*)p;   p += alignup((size_t)MH_DIM * MH_DIM * 2);

    const int T = 256;
    const int nf = N_NODES * F_DIM;
    const int scatterT = N_EDGES * 32;

    k_cvt_bf16<<<(nf + T - 1) / T, T, 0, stream>>>(x,   xb,   nf);
    k_cvt_bf16<<<(F_DIM*F_DIM + T - 1) / T, T, 0, stream>>>(W1l, W1lb, F_DIM*F_DIM);
    k_cvt_bf16<<<(F_DIM*F_DIM + T - 1) / T, T, 0, stream>>>(W1r, W1rb, F_DIM*F_DIM);
    k_cvt_bf16<<<(F_DIM*F_DIM + T - 1) / T, T, 0, stream>>>(W2l, W2lb, F_DIM*F_DIM);
    k_cvt_bf16<<<(F_DIM*F_DIM + T - 1) / T, T, 0, stream>>>(W2r, W2rb, F_DIM*F_DIM);
    k_cvt_bf16<<<(MH_DIM*MH_DIM + T - 1) / T, T, 0, stream>>>(Wm1, Wm1b, MH_DIM*MH_DIM);

    // ---- layer 1 ----
    k_zero_f32<<<(N_NODES + T - 1) / T, T, 0, stream>>>(cnt, N_NODES);
    k_zero_f32<<<(nf + T - 1) / T, T, 0, stream>>>(agg, nf);
    k_scatter_f32<<<(scatterT + T - 1) / T, T, 0, stream>>>(x, srcI, dstI, agg, cnt);
    k_mean_bf16<<<(nf + T - 1) / T, T, 0, stream>>>(agg, cnt, aggb);
    k_sage_wmma<<<(N_NODES / 16) * (F_DIM / 16) / 8, T, 0, stream>>>(
        aggb, xb, W1lb, W1rb, b1l, hb1);

    // ---- layer 2 ----
    k_zero_f32<<<(nf + T - 1) / T, T, 0, stream>>>(agg, nf);
    k_scatter_bf16<<<(scatterT + T - 1) / T, T, 0, stream>>>(hb1, srcI, dstI, agg);
    k_mean_bf16<<<(nf + T - 1) / T, T, 0, stream>>>(agg, cnt, aggb);
    k_sage_wmma<<<(N_NODES / 16) * (F_DIM / 16) / 8, T, 0, stream>>>(
        aggb, hb1, W2lb, W2rb, b2l, hb2);

    // ---- edge MLP: TDM-staged weights in LDS, 132KB dynamic shared ----
    k_edge_mlp<<<(N_EDGES / 32) / 8, T, (size_t)(MH_DIM * WROW * 2), stream>>>(
        hb2, srcI, dstI, Wm1b, bm1, Wm2, bm2, out);
}
